// mLSTM_28226525070215
// MI455X (gfx1250) — matrix operand order repaired and checked by reading, not compiled, not run on hardware
//
#include <hip/hip_runtime.h>
#include <hip/hip_bf16.h>
#include <math.h>

// mLSTM final-state kernel for gfx1250 (MI455X).
// Pipeline: conv1d -> bf16 pack -> 4x WMMA GEMM (k,v,i,f projections, fused
// activations) -> per-column suffix-product scan -> a_t = g_t . q_T ->
// h = o * (V^T a) / max(|sum a|, 1).

typedef __attribute__((ext_vector_type(16))) __bf16 v16bf;
typedef __attribute__((ext_vector_type(8)))  float  v8f;

#define T_SEQ 2048
#define TP    2051          // T + 3 conv output rows
#define TPAD  2064          // 129 * 16 (padded M for WMMA tiles)
#define F_IN  256           // K of the projection GEMMs
#define H_DIM 384
#define NTOT  1536          // 4 * H  (columns: [k | v | i | f])
#define MTILES 129
#define NTILES 96

// ---- workspace layout (bytes), all 256B-aligned ----
#define OFF_XPF 0u                                  // float  [TPAD*F_IN]
#define OFF_XPB (OFF_XPF + TPAD*F_IN*4u)            // ushort [TPAD*F_IN]
#define OFF_WB  (OFF_XPB + TPAD*F_IN*2u)            // ushort [NTOT*F_IN]
#define OFF_P   (OFF_WB  + NTOT*F_IN*2u)            // float  [TPAD*NTOT]
#define OFF_G   (OFF_P   + (size_t)TPAD*NTOT*4u)    // float  [TPAD*H_DIM]
#define OFF_A   (OFF_G   + (size_t)TPAD*H_DIM*4u)   // float  [TPAD]
#define OFF_Q   (OFF_A   + TPAD*4u + 192u)          // float  [H_DIM]
#define OFF_O   (OFF_Q   + H_DIM*4u)                // float  [H_DIM]

__device__ __forceinline__ unsigned short f2bf(float x) {
    union { float f; unsigned u; } c; c.f = x;
    unsigned r = (c.u + 0x7FFFu + ((c.u >> 16) & 1u)) >> 16;
    return (unsigned short)r;
}

// Pack [Wk;Wv;Wi;Wf] (each H_DIM x F_IN, row-major) into one bf16 B matrix.
__global__ void pack_w_kernel(const float* __restrict__ Wk, const float* __restrict__ Wv,
                              const float* __restrict__ Wi, const float* __restrict__ Wf,
                              unsigned short* __restrict__ WB) {
    int idx = blockIdx.x * blockDim.x + threadIdx.x;
    if (idx >= NTOT * F_IN) return;
    int g = idx / (H_DIM * F_IN);
    int rem = idx - g * (H_DIM * F_IN);
    const float* src = (g == 0) ? Wk : (g == 1) ? Wv : (g == 2) ? Wi : Wf;
    WB[idx] = f2bf(src[rem]);
}

// Causal-ish conv1d(kernel=4, pad=(3,3)) over time per feature; writes fp32 and
// bf16 copies of xp, zero-filling the pad rows [TP, TPAD).
__global__ void conv_kernel(const float* __restrict__ x, const float* __restrict__ cw,
                            const float* __restrict__ cb,
                            float* __restrict__ xpf, unsigned short* __restrict__ xpb) {
    int idx = blockIdx.x * blockDim.x + threadIdx.x;
    if (idx >= TPAD * F_IN) return;
    int t = idx / F_IN, f = idx - t * F_IN;
    float acc = 0.f;
    if (t < TP) {
        acc = cb[0];
        #pragma unroll
        for (int k = 0; k < 4; ++k) {
            int ts = t + k - 3;
            if (ts >= 0 && ts < T_SEQ) acc += cw[k] * x[ts * F_IN + f];
        }
    }
    xpf[idx] = acc;
    xpb[idx] = f2bf(acc);
}

// One 16x16 output tile per wave; K-loop of 8 x v_wmma_f32_16x16x32_bf16.
// Fused epilogue: bias + per-group activation (k: *1/sqrt(H), v: id, i: exp,
// f: sigmoid). 384 % 16 == 0 so group is wave-uniform.
__global__ void __launch_bounds__(128)
gemm_kernel(const unsigned short* __restrict__ A, const unsigned short* __restrict__ B,
            const float* __restrict__ bk, const float* __restrict__ bv,
            const float* __restrict__ bi, const float* __restrict__ bf,
            float* __restrict__ P) {
    int wave = blockIdx.x * 4 + (threadIdx.x >> 5);
    int lane = threadIdx.x & 31;
    int tm = wave % MTILES, tn = wave / MTILES;
    if (tn >= NTILES) return;
    int m0 = tm * 16, n0 = tn * 16;
    int half = lane >> 4, r = lane & 15;

    const unsigned short* arow = A + (size_t)(m0 + r) * F_IN;
    const unsigned short* brow = B + (size_t)(n0 + r) * F_IN;

    v8f acc = {0.f, 0.f, 0.f, 0.f, 0.f, 0.f, 0.f, 0.f};
    #pragma unroll
    for (int kk = 0; kk < F_IN; kk += 32) {
        union { v16bf v; uint4 q[2]; } a, b;
        // A 16x32 bf16: lane half selects K {0..7,16..23} vs {8..15,24..31}
        a.q[0] = *(const uint4*)(arow + kk + half * 8);
        a.q[1] = *(const uint4*)(arow + kk + 16 + half * 8);
        // B 32x16 bf16 (N-major = W row-major): lane half selects K {0..7,16..23} vs {8..15,24..31}
        b.q[0] = *(const uint4*)(brow + kk + half * 8);
        b.q[1] = *(const uint4*)(brow + kk + 16 + half * 8);
        acc = __builtin_amdgcn_wmma_f32_16x16x32_bf16(false, a.v, false, b.v,
                                                      (short)0, acc, false, false);
    }

    int grp = n0 / H_DIM;
    const float* bptr = (grp == 0) ? bk : (grp == 1) ? bv : (grp == 2) ? bi : bf;
    float bias = bptr[(n0 % H_DIM) + r];
    const float inv_sqrt_h = 0.05103103630798288f; // 1/sqrt(384)
    #pragma unroll
    for (int e = 0; e < 8; ++e) {
        int row = m0 + e + half * 8;
        float val = acc[e] + bias;
        if (grp == 0)      val *= inv_sqrt_h;
        else if (grp == 2) val = __expf(val);
        else if (grp == 3) val = 1.f / (1.f + __expf(-val));
        P[(size_t)row * NTOT + n0 + r] = val;
    }
}

// Per-column suffix scan: g_t[j] = (prod_{s>t} f_s[j]) * i_t[j] * k_t[j].
__global__ void scan_kernel(const float* __restrict__ P, float* __restrict__ G) {
    int j = blockIdx.x * blockDim.x + threadIdx.x;
    if (j >= H_DIM) return;
    float sp = 1.f;
    for (int t = TP - 1; t >= 0; --t) {
        const float* row = P + (size_t)t * NTOT;
        float kv = row[j];
        float iv = row[2 * H_DIM + j];
        float fv = row[3 * H_DIM + j];
        G[(size_t)t * H_DIM + j] = sp * iv * kv;
        sp *= fv;
    }
}

// q_T and o_T from the final xp row only (tiny matvecs).
__global__ void qo_kernel(const float* __restrict__ xpf,
                          const float* __restrict__ Wq, const float* __restrict__ bq,
                          const float* __restrict__ Wo, const float* __restrict__ bo,
                          float* __restrict__ qv, float* __restrict__ ov) {
    int idx = blockIdx.x * blockDim.x + threadIdx.x;
    if (idx >= 2 * H_DIM) return;
    int n = idx % H_DIM;
    const float* xrow = xpf + (size_t)(TP - 1) * F_IN;
    const float* W = (idx < H_DIM) ? (Wq + (size_t)n * F_IN) : (Wo + (size_t)n * F_IN);
    float acc = (idx < H_DIM) ? bq[n] : bo[n];
    for (int k = 0; k < F_IN; ++k) acc += xrow[k] * W[k];
    if (idx < H_DIM) qv[n] = acc;
    else             ov[n] = 1.f / (1.f + __expf(-acc));
}

// a_t = g_t . q_T  (one wave32 per time step).
__global__ void a_kernel(const float* __restrict__ G, const float* __restrict__ qv,
                         float* __restrict__ av) {
    int wave = blockIdx.x * (blockDim.x >> 5) + (threadIdx.x >> 5);
    int lane = threadIdx.x & 31;
    if (wave >= TP) return;
    const float* grow = G + (size_t)wave * H_DIM;
    float s = 0.f;
    for (int j = lane; j < H_DIM; j += 32) s += grow[j] * qv[j];
    for (int off = 16; off; off >>= 1) s += __shfl_down(s, off, 32);
    if (lane == 0) av[wave] = s;
}

// h[r] = o[r] * (sum_t v_t[r] * a_t) / max(|sum_t a_t|, 1)
__global__ void final_kernel(const float* __restrict__ P, const float* __restrict__ av,
                             const float* __restrict__ ov, float* __restrict__ out) {
    __shared__ float sm[H_DIM];
    int r = threadIdx.x;
    float s = 0.f;
    for (int t = r; t < TP; t += H_DIM) s += av[t];
    sm[r] = s;
    __syncthreads();
    if (r < 128) sm[r] += sm[r + 256];          // fold 384 -> 256
    __syncthreads();
    for (int st = 128; st > 0; st >>= 1) {
        if (r < st) sm[r] += sm[r + st];
        __syncthreads();
    }
    float den = fmaxf(fabsf(sm[0]), 1.f);
    float cq = 0.f;
    for (int t = 0; t < TP; ++t)                 // v group at column offset H_DIM
        cq += P[(size_t)t * NTOT + H_DIM + r] * av[t];
    out[r] = ov[r] * cq / den;
}

extern "C" void kernel_launch(void* const* d_in, const int* in_sizes, int n_in,
                              void* d_out, int out_size, void* d_ws, size_t ws_size,
                              hipStream_t stream) {
    const float* x  = (const float*)d_in[0];
    const float* Wq = (const float*)d_in[1];
    const float* bq = (const float*)d_in[2];
    const float* Wk = (const float*)d_in[3];
    const float* bk = (const float*)d_in[4];
    const float* Wv = (const float*)d_in[5];
    const float* bv = (const float*)d_in[6];
    const float* Wi = (const float*)d_in[7];
    const float* bi = (const float*)d_in[8];
    const float* Wf = (const float*)d_in[9];
    const float* bf = (const float*)d_in[10];
    const float* Wo = (const float*)d_in[11];
    const float* bo = (const float*)d_in[12];
    const float* cw = (const float*)d_in[13];
    const float* cb = (const float*)d_in[14];

    char* ws = (char*)d_ws;
    float*          xpf = (float*)(ws + OFF_XPF);
    unsigned short* xpb = (unsigned short*)(ws + OFF_XPB);
    unsigned short* WB  = (unsigned short*)(ws + OFF_WB);
    float*          P   = (float*)(ws + OFF_P);
    float*          G   = (float*)(ws + OFF_G);
    float*          av  = (float*)(ws + OFF_A);
    float*          qv  = (float*)(ws + OFF_Q);
    float*          ov  = (float*)(ws + OFF_O);

    pack_w_kernel<<<(NTOT * F_IN + 255) / 256, 256, 0, stream>>>(Wk, Wv, Wi, Wf, WB);
    conv_kernel<<<(TPAD * F_IN + 255) / 256, 256, 0, stream>>>(x, cw, cb, xpf, xpb);
    gemm_kernel<<<(MTILES * NTILES) / 4, 128, 0, stream>>>(xpb, WB, bk, bv, bi, bf, P);
    scan_kernel<<<(H_DIM + 127) / 128, 128, 0, stream>>>(P, G);
    qo_kernel<<<(2 * H_DIM + 255) / 256, 256, 0, stream>>>(xpf, Wq, bq, Wo, bo, qv, ov);
    a_kernel<<<(TP + 7) / 8, 256, 0, stream>>>(G, qv, av);
    final_kernel<<<1, H_DIM, 0, stream>>>(P, av, ov, (float*)d_out);
}
